// TransformerBaseBlock_6751688589649
// MI455X (gfx1250) — matrix operand order repaired and checked by reading, not compiled, not run on hardware
//
#include <hip/hip_runtime.h>

// ---------------------------------------------------------------------------
// Transformer base block for MI455X (gfx1250), bf16 WMMA pipeline.
// D=256, HEADS=4, DEPTH=64, EXPAND=4 (hidden 1024), B=4, S=2048, EPS=1e-3.
// ---------------------------------------------------------------------------

#define TB_DIM   256
#define TB_SEQ   2048
#define TB_NH    4
#define TB_DEPTH 64
#define TB_HID   1024
#define TB_ROWS  (4 * TB_SEQ)   // 8192
#define TB_EPS   1e-3f

typedef __attribute__((ext_vector_type(16))) __bf16        v16bf;
typedef __attribute__((ext_vector_type(8)))  float         v8f;
typedef __attribute__((ext_vector_type(4)))  unsigned int  u32x4;

union TbFrag { u32x4 q[2]; v16bf v; };

// A-fragment and B-fragment: two 16B chunks (base, base+16 halves)
#define TB_LDA(frag, ptr) do { (frag).q[0] = *(const u32x4*)(ptr);  \
                               (frag).q[1] = *(const u32x4*)((ptr) + 16); } while (0)
#define TB_LDB(frag, ptr) do { (frag).q[0] = *(const u32x4*)(ptr);  \
                               (frag).q[1] = *(const u32x4*)((ptr) + 16); } while (0)

static __device__ __forceinline__ unsigned short tb_f2bf(float x) {
    union { float f; unsigned int u; } c; c.f = x;
    unsigned int r = c.u + 0x7FFFu + ((c.u >> 16) & 1u);   // round-nearest-even
    return (unsigned short)(r >> 16);
}

// ---------------------------------------------------------------------------
// Prep kernels
// ---------------------------------------------------------------------------

// WT[n][k] = bf16(W[k][n]) ; W is K x N row-major f32
__global__ __launch_bounds__(256) void tb_wt_transpose(
    const float* __restrict__ W, unsigned short* __restrict__ WT, int K, int N) {
    int idx = blockIdx.x * 256 + threadIdx.x;
    if (idx >= K * N) return;
    int n = idx % N;
    int k = idx / N;
    WT[(size_t)n * K + k] = tb_f2bf(W[(size_t)k * N + n]);
}

// h = bf16( bn1(x) ), elementwise over 8192 x 256
__global__ __launch_bounds__(256) void tb_bn1(
    const float* __restrict__ x, const float* __restrict__ g,
    const float* __restrict__ be, const float* __restrict__ mu,
    const float* __restrict__ var, unsigned short* __restrict__ h) {
    int idx = blockIdx.x * 256 + threadIdx.x;
    if (idx >= TB_ROWS * TB_DIM) return;
    int c = idx & (TB_DIM - 1);
    float v = g[c] * (x[idx] - mu[c]) * rsqrtf(var[c] + TB_EPS) + be[c];
    h[idx] = tb_f2bf(v);
}

// ---------------------------------------------------------------------------
// WMMA GEMM core: one wave computes a 16x64 f32 tile of  A[R,kdim] * W[kdim,N]
// with W stored transposed bf16 (WT[n][k], row stride = kdim).
// Ping-pong double buffering over a 2x-unrolled k-loop (kdim % 64 == 0):
// no register rotation copies, loads stay one half-step ahead of the WMMAs.
// ---------------------------------------------------------------------------
static __device__ __forceinline__ void tb_gemm_tile(
    const unsigned short* __restrict__ A,
    const unsigned short* __restrict__ WT,
    int kdim, int row0, int col0, v8f acc[4]) {
    const int lane  = threadIdx.x & 31;
    const int m     = lane & 15;
    const int koffA = (lane < 16) ? 0 : 8;    // A-matrix 16-bit layout (ISA 7.12.2)
    const int koffB = (lane < 16) ? 0 : 8;    // B-matrix 16-bit layout
    const unsigned short* arow = A + (size_t)(row0 + m) * kdim + koffA;
    const unsigned short* wrow = WT + (size_t)(col0 + m) * kdim + koffB;

    TbFrag a0, b0[4], a1, b1[4];
    TB_LDA(a0, arow);
#pragma unroll
    for (int t = 0; t < 4; ++t) TB_LDB(b0[t], wrow + (size_t)(t * 16) * kdim);

    for (int kb = 0; kb < kdim; kb += 64) {
        // stage set1 @ kb+32 while set0 computes
        TB_LDA(a1, arow + kb + 32);
#pragma unroll
        for (int t = 0; t < 4; ++t) TB_LDB(b1[t], wrow + (size_t)(t * 16) * kdim + kb + 32);
        __builtin_prefetch(arow + kb + 128, 0, 1);   // global_prefetch_b8
#pragma unroll
        for (int t = 0; t < 4; ++t)
            acc[t] = __builtin_amdgcn_wmma_f32_16x16x32_bf16(
                false, a0.v, false, b0[t].v, (short)0, acc[t], false, false);
        // stage set0 @ kb+64 while set1 computes
        if (kb + 64 < kdim) {
            TB_LDA(a0, arow + kb + 64);
#pragma unroll
            for (int t = 0; t < 4; ++t) TB_LDB(b0[t], wrow + (size_t)(t * 16) * kdim + kb + 64);
        }
#pragma unroll
        for (int t = 0; t < 4; ++t)
            acc[t] = __builtin_amdgcn_wmma_f32_16x16x32_bf16(
                false, a1.v, false, b1[t].v, (short)0, acc[t], false, false);
    }
}

// ---- epilogue variant: bias + bf16 row-major (Q / K projections) ----------
__global__ __launch_bounds__(256) void tb_gemm_bias_bf16(
    const unsigned short* __restrict__ A, const unsigned short* __restrict__ WT,
    const float* __restrict__ bias, unsigned short* __restrict__ out,
    int kdim, int ldo) {
    const int wid  = threadIdx.x >> 5;
    const int lane = threadIdx.x & 31;
    const int m    = lane & 15;
    const int hi   = (lane < 16) ? 0 : 8;
    const int row0 = blockIdx.x * 128 + wid * 16;
    const int col0 = blockIdx.y * 64;
    v8f acc[4] = { (v8f)0.f, (v8f)0.f, (v8f)0.f, (v8f)0.f };
    tb_gemm_tile(A, WT, kdim, row0, col0, acc);
#pragma unroll
    for (int t = 0; t < 4; ++t) {
        int c = col0 + t * 16 + m;
        float bn = bias[c];
#pragma unroll
        for (int r = 0; r < 8; ++r)
            out[(size_t)(row0 + r + hi) * ldo + c] = tb_f2bf(acc[t][r] + bn);
    }
}

// ---- epilogue variant: bias + transposed store vT[((b*H+h)*64+dl)*S + s] ---
__global__ __launch_bounds__(256) void tb_gemm_bias_vT(
    const unsigned short* __restrict__ A, const unsigned short* __restrict__ WT,
    const float* __restrict__ bias, unsigned short* __restrict__ vT) {
    const int wid  = threadIdx.x >> 5;
    const int lane = threadIdx.x & 31;
    const int m    = lane & 15;
    const int hi   = (lane < 16) ? 0 : 8;
    const int row0 = blockIdx.x * 128 + wid * 16;
    const int col0 = blockIdx.y * 64;
    v8f acc[4] = { (v8f)0.f, (v8f)0.f, (v8f)0.f, (v8f)0.f };
    tb_gemm_tile(A, WT, TB_DIM, row0, col0, acc);
    const int b = row0 / TB_SEQ;
#pragma unroll
    for (int t = 0; t < 4; ++t) {
        int c = col0 + t * 16 + m;
        float bn = bias[c];
        int h  = c >> 6;         // DEPTH=64
        int dl = c & 63;
        unsigned short* dst = vT + (size_t)((b * TB_NH + h) * TB_DEPTH + dl) * TB_SEQ;
#pragma unroll
        for (int r = 0; r < 8; ++r) {
            int s = row0 - b * TB_SEQ + r + hi;
            dst[s] = tb_f2bf(acc[t][r] + bn);
        }
    }
}

// ---- epilogue variant: Wo proj + bo + residual x -> x1(f32), BN2 -> h2(bf16)
__global__ __launch_bounds__(256) void tb_gemm_wo(
    const unsigned short* __restrict__ A, const unsigned short* __restrict__ WT,
    const float* __restrict__ bo, const float* __restrict__ x,
    const float* __restrict__ g2, const float* __restrict__ b2,
    const float* __restrict__ mu2, const float* __restrict__ var2,
    float* __restrict__ x1, unsigned short* __restrict__ h2) {
    const int wid  = threadIdx.x >> 5;
    const int lane = threadIdx.x & 31;
    const int m    = lane & 15;
    const int hi   = (lane < 16) ? 0 : 8;
    const int row0 = blockIdx.x * 128 + wid * 16;
    const int col0 = blockIdx.y * 64;
    v8f acc[4] = { (v8f)0.f, (v8f)0.f, (v8f)0.f, (v8f)0.f };
    tb_gemm_tile(A, WT, TB_DIM, row0, col0, acc);
#pragma unroll
    for (int t = 0; t < 4; ++t) {
        int c = col0 + t * 16 + m;
        float bn = bo[c];
        float gg = g2[c], bb = b2[c], mm = mu2[c];
        float iv = rsqrtf(var2[c] + TB_EPS);
#pragma unroll
        for (int r = 0; r < 8; ++r) {
            size_t idx = (size_t)(row0 + r + hi) * TB_DIM + c;
            float v = acc[t][r] + bn + x[idx];
            x1[idx] = v;
            h2[idx] = tb_f2bf(gg * (v - mm) * iv + bb);
        }
    }
}

// ---- epilogue variant: exact GELU -> bf16 (FFN up-projection) -------------
__global__ __launch_bounds__(256) void tb_gemm_gelu(
    const unsigned short* __restrict__ A, const unsigned short* __restrict__ WT,
    unsigned short* __restrict__ out, int kdim, int ldo) {
    const int wid  = threadIdx.x >> 5;
    const int lane = threadIdx.x & 31;
    const int m    = lane & 15;
    const int hi   = (lane < 16) ? 0 : 8;
    const int row0 = blockIdx.x * 128 + wid * 16;
    const int col0 = blockIdx.y * 64;
    v8f acc[4] = { (v8f)0.f, (v8f)0.f, (v8f)0.f, (v8f)0.f };
    tb_gemm_tile(A, WT, kdim, row0, col0, acc);
#pragma unroll
    for (int t = 0; t < 4; ++t) {
        int c = col0 + t * 16 + m;
#pragma unroll
        for (int r = 0; r < 8; ++r) {
            float v = acc[t][r];
            float ge = 0.5f * v * (1.0f + erff(v * 0.70710678118654752f));
            out[(size_t)(row0 + r + hi) * ldo + c] = tb_f2bf(ge);
        }
    }
}

// ---- epilogue variant: + x1 residual -> f32 final output ------------------
__global__ __launch_bounds__(256) void tb_gemm_resid(
    const unsigned short* __restrict__ A, const unsigned short* __restrict__ WT,
    const float* __restrict__ x1, float* __restrict__ out, int kdim) {
    const int wid  = threadIdx.x >> 5;
    const int lane = threadIdx.x & 31;
    const int m    = lane & 15;
    const int hi   = (lane < 16) ? 0 : 8;
    const int row0 = blockIdx.x * 128 + wid * 16;
    const int col0 = blockIdx.y * 64;
    v8f acc[4] = { (v8f)0.f, (v8f)0.f, (v8f)0.f, (v8f)0.f };
    tb_gemm_tile(A, WT, kdim, row0, col0, acc);
#pragma unroll
    for (int t = 0; t < 4; ++t) {
        int c = col0 + t * 16 + m;
#pragma unroll
        for (int r = 0; r < 8; ++r) {
            size_t idx = (size_t)(row0 + r + hi) * TB_DIM + c;
            out[idx] = acc[t][r] + x1[idx];
        }
    }
}

// ---------------------------------------------------------------------------
// Flash attention: one wave = 16 queries of one (b,h); streams 64 keys/step.
// K fragments are batch-loaded before the 8 score WMMAs; V fragments are
// issued before the softmax VALU work so their latency hides behind it.
// P is transposed C-layout -> A-layout through LDS (s_wait_dscnt 0).
// ---------------------------------------------------------------------------
__global__ __launch_bounds__(256) void tb_attention(
    const unsigned short* __restrict__ qb, const unsigned short* __restrict__ kbm,
    const unsigned short* __restrict__ vT, unsigned short* __restrict__ ctx) {
    const int lane  = threadIdx.x & 31;
    const int wid   = threadIdx.x >> 5;
    const int m     = lane & 15;
    const int hi    = (lane < 16) ? 0 : 8;
    const int koffA = (lane < 16) ? 0 : 8;
    const int koffB = (lane < 16) ? 0 : 8;

    const int bh = blockIdx.y;            // b*NH + h, 0..15
    const int b  = bh >> 2;
    const int h  = bh & 3;
    const int q0 = blockIdx.x * 128 + wid * 16;

    __shared__ alignas(16) unsigned short Pl[8][16 * 64];   // per-wave P tile

    // Q A-fragments over depth 64 (two 32-wide K chunks)
    const unsigned short* qrow =
        qb + (size_t)(b * TB_SEQ + q0 + m) * TB_DIM + h * TB_DEPTH + koffA;
    TbFrag aq0, aq1;
    TB_LDA(aq0, qrow);
    TB_LDA(aq1, qrow + 32);

    float mrow[8], lrow[8];
    v8f o[4] = { (v8f)0.f, (v8f)0.f, (v8f)0.f, (v8f)0.f };
#pragma unroll
    for (int r = 0; r < 8; ++r) { mrow[r] = -3.0e38f; lrow[r] = 0.f; }

    const unsigned short* kbase = kbm + (size_t)(b * TB_SEQ) * TB_DIM + h * TB_DEPTH + koffB;

    for (int kt = 0; kt < TB_SEQ; kt += 64) {
        // ---- batch-load all K fragments for this 64-key step ----------------
        TbFrag bk[8];
#pragma unroll
        for (int j = 0; j < 4; ++j) {
            const unsigned short* krow = kbase + (size_t)(kt + j * 16 + m) * TB_DIM;
            __builtin_prefetch(krow + 64 * TB_DIM, 0, 1);
            TB_LDB(bk[2 * j],     krow);
            TB_LDB(bk[2 * j + 1], krow + 32);
        }
        // ---- 8 score WMMAs back-to-back -------------------------------------
        v8f s[4];
#pragma unroll
        for (int j = 0; j < 4; ++j) {
            v8f a = (v8f)0.f;
            a = __builtin_amdgcn_wmma_f32_16x16x32_bf16(false, aq0.v, false, bk[2 * j].v,     (short)0, a, false, false);
            a = __builtin_amdgcn_wmma_f32_16x16x32_bf16(false, aq1.v, false, bk[2 * j + 1].v, (short)0, a, false, false);
            s[j] = a * 0.125f;   // 1/sqrt(64)
        }
        // ---- issue V fragment loads now; they complete during the softmax ---
        TbFrag bv[8];
#pragma unroll
        for (int t = 0; t < 4; ++t) {
            const unsigned short* vrow =
                vT + (size_t)(bh * TB_DEPTH + t * 16 + m) * TB_SEQ + kt + koffB;
            TB_LDB(bv[2 * t],     vrow);
            TB_LDB(bv[2 * t + 1], vrow + 32);
        }
        // ---- online softmax, write P (bf16) to LDS in matrix order ----------
#pragma unroll
        for (int r = 0; r < 8; ++r) {
            float v0 = s[0][r], v1 = s[1][r], v2 = s[2][r], v3 = s[3][r];
            float tm = fmaxf(fmaxf(v0, v1), fmaxf(v2, v3));
            tm = fmaxf(tm, __shfl_xor(tm, 1, 32));
            tm = fmaxf(tm, __shfl_xor(tm, 2, 32));
            tm = fmaxf(tm, __shfl_xor(tm, 4, 32));
            tm = fmaxf(tm, __shfl_xor(tm, 8, 32));
            float nm = fmaxf(mrow[r], tm);
            float al = __expf(mrow[r] - nm);
            float p0 = __expf(v0 - nm), p1 = __expf(v1 - nm);
            float p2 = __expf(v2 - nm), p3 = __expf(v3 - nm);
            float ls = (p0 + p1) + (p2 + p3);
            ls += __shfl_xor(ls, 1, 32);
            ls += __shfl_xor(ls, 2, 32);
            ls += __shfl_xor(ls, 4, 32);
            ls += __shfl_xor(ls, 8, 32);
            lrow[r] = lrow[r] * al + ls;
            mrow[r] = nm;
            o[0][r] *= al; o[1][r] *= al; o[2][r] *= al; o[3][r] *= al;
            unsigned short* pp = &Pl[wid][(r + hi) * 64 + m];
            pp[0]  = tb_f2bf(p0);
            pp[16] = tb_f2bf(p1);
            pp[32] = tb_f2bf(p2);
            pp[48] = tb_f2bf(p3);
        }
        asm volatile("s_wait_dscnt 0" ::: "memory");
        // ---- reload P in A-matrix layout ------------------------------------
        const unsigned short* prow = &Pl[wid][m * 64 + koffA];
        TbFrag pa0, pa1;
        TB_LDA(pa0, prow);
        TB_LDA(pa1, prow + 32);
        // ---- 8 PV WMMAs back-to-back ----------------------------------------
#pragma unroll
        for (int t = 0; t < 4; ++t) {
            o[t] = __builtin_amdgcn_wmma_f32_16x16x32_bf16(false, pa0.v, false, bv[2 * t].v,     (short)0, o[t], false, false);
            o[t] = __builtin_amdgcn_wmma_f32_16x16x32_bf16(false, pa1.v, false, bv[2 * t + 1].v, (short)0, o[t], false, false);
        }
    }
    // ---- normalize and store context (row-major, col = h*64 + d) ------------
#pragma unroll
    for (int r = 0; r < 8; ++r) {
        float inv = 1.0f / lrow[r];
        size_t row = (size_t)(b * TB_SEQ + q0 + r + hi);
#pragma unroll
        for (int t = 0; t < 4; ++t)
            ctx[row * TB_DIM + h * TB_DEPTH + t * 16 + m] = tb_f2bf(o[t][r] * inv);
    }
}

// ---------------------------------------------------------------------------
// Host-side launch
// ---------------------------------------------------------------------------
extern "C" void kernel_launch(void* const* d_in, const int* in_sizes, int n_in,
                              void* d_out, int out_size, void* d_ws, size_t ws_size,
                              hipStream_t stream) {
    (void)in_sizes; (void)n_in; (void)out_size; (void)ws_size;
    const float* x    = (const float*)d_in[0];
    const float* g1   = (const float*)d_in[1];
    const float* be1  = (const float*)d_in[2];
    const float* mu1  = (const float*)d_in[3];
    const float* var1 = (const float*)d_in[4];
    const float* wq   = (const float*)d_in[5];
    const float* bq   = (const float*)d_in[6];
    const float* wk   = (const float*)d_in[7];
    const float* bk   = (const float*)d_in[8];
    const float* wv   = (const float*)d_in[9];
    const float* bv   = (const float*)d_in[10];
    const float* wo   = (const float*)d_in[11];
    const float* bo   = (const float*)d_in[12];
    const float* g2   = (const float*)d_in[13];
    const float* be2  = (const float*)d_in[14];
    const float* mu2  = (const float*)d_in[15];
    const float* var2 = (const float*)d_in[16];
    const float* w1   = (const float*)d_in[17];
    const float* w2   = (const float*)d_in[18];

    char* base = (char*)d_ws;
    size_t off = 0;
    auto alloc = [&](size_t bytes) -> void* {
        void* r = base + off;
        off += (bytes + 255) & ~(size_t)255;
        return r;
    };
    const size_t act = (size_t)TB_ROWS * TB_DIM;      // 8192*256
    unsigned short* wqT = (unsigned short*)alloc(TB_DIM * TB_DIM * 2);
    unsigned short* wkT = (unsigned short*)alloc(TB_DIM * TB_DIM * 2);
    unsigned short* wvT = (unsigned short*)alloc(TB_DIM * TB_DIM * 2);
    unsigned short* woT = (unsigned short*)alloc(TB_DIM * TB_DIM * 2);
    unsigned short* w1T = (unsigned short*)alloc((size_t)TB_DIM * TB_HID * 2);
    unsigned short* w2T = (unsigned short*)alloc((size_t)TB_HID * TB_DIM * 2);
    unsigned short* hbf = (unsigned short*)alloc(act * 2);
    unsigned short* qbf = (unsigned short*)alloc(act * 2);
    unsigned short* kbf = (unsigned short*)alloc(act * 2);
    unsigned short* vTb = (unsigned short*)alloc(act * 2);
    unsigned short* ctx = (unsigned short*)alloc(act * 2);
    unsigned short* h2b = (unsigned short*)alloc(act * 2);
    float*          x1  = (float*)alloc(act * 4);
    unsigned short* m1  = (unsigned short*)alloc((size_t)TB_ROWS * TB_HID * 2);

    dim3 blk(256);
    // weight prep (f32 -> bf16, transposed)
    tb_wt_transpose<<<(TB_DIM * TB_DIM + 255) / 256, blk, 0, stream>>>(wq, wqT, TB_DIM, TB_DIM);
    tb_wt_transpose<<<(TB_DIM * TB_DIM + 255) / 256, blk, 0, stream>>>(wk, wkT, TB_DIM, TB_DIM);
    tb_wt_transpose<<<(TB_DIM * TB_DIM + 255) / 256, blk, 0, stream>>>(wv, wvT, TB_DIM, TB_DIM);
    tb_wt_transpose<<<(TB_DIM * TB_DIM + 255) / 256, blk, 0, stream>>>(wo, woT, TB_DIM, TB_DIM);
    tb_wt_transpose<<<(TB_DIM * TB_HID + 255) / 256, blk, 0, stream>>>(w1, w1T, TB_DIM, TB_HID);
    tb_wt_transpose<<<(TB_HID * TB_DIM + 255) / 256, blk, 0, stream>>>(w2, w2T, TB_HID, TB_DIM);
    // BN1
    tb_bn1<<<(int)(act / 256), blk, 0, stream>>>(x, g1, be1, mu1, var1, hbf);
    // QKV projections
    dim3 gD(TB_ROWS / 128, TB_DIM / 64);
    tb_gemm_bias_bf16<<<gD, blk, 0, stream>>>(hbf, wqT, bq, qbf, TB_DIM, TB_DIM);
    tb_gemm_bias_bf16<<<gD, blk, 0, stream>>>(hbf, wkT, bk, kbf, TB_DIM, TB_DIM);
    tb_gemm_bias_vT  <<<gD, blk, 0, stream>>>(hbf, wvT, bv, vTb);
    // attention
    tb_attention<<<dim3(TB_SEQ / 128, 16), blk, 0, stream>>>(qbf, kbf, vTb, ctx);
    // output projection + residual + BN2
    tb_gemm_wo<<<gD, blk, 0, stream>>>(ctx, woT, bo, x, g2, be2, mu2, var2, x1, h2b);
    // FFN
    dim3 gH(TB_ROWS / 128, TB_HID / 64);
    tb_gemm_gelu <<<gH, blk, 0, stream>>>(h2b, w1T, m1, TB_DIM, TB_HID);
    tb_gemm_resid<<<gD, blk, 0, stream>>>(m1, w2T, x1, (float*)d_out, TB_HID);
}
